// Hexagon108Conv2d_20633022890659
// MI455X (gfx1250) — matrix operand order repaired and checked by reading, not compiled, not run on hardware
//
#include <hip/hip_runtime.h>

// ---------------------------------------------------------------------------
// Hexagonal conv 13x13, Cin=Cout=64, B=4096 — tap-decomposed implicit GEMM.
//   y[co,n] = OUT_MASK * (bias + sum_{t in 7 hex taps} W_t[co,:] . X[:, n+shift(t)])
// 7 GEMMs of K=64 (2 WMMA K-steps each), bf16 in, fp32 accumulate.
// Loop nest: (tap,K) outer, N inner — each A fragment loaded ONCE per wave,
// 6 accumulators (96 N-columns) held live. B-fragments are contiguous
// ds_load_b128 from a channel-innermost LDS image; all tap/K offsets are
// compile-time immediates (no inner-loop VALU).
// ---------------------------------------------------------------------------

typedef __attribute__((ext_vector_type(16))) __bf16 v16bf;
typedef __attribute__((ext_vector_type(8)))  float  v8f;

struct __attribute__((aligned(16))) U4 { unsigned int x, y, z, w; };
union Frag { U4 q[2]; v16bf v; };

#define B_TOTAL 4096
#define CIN     64
#define COUT    64
#define HW      13
#define NPIX    169
#define HP      15
#define PSTRIDE 72          // u16 per padded-pixel slot: 64 ci + 8 pad (16B mult)
#define NTAPS   7
#define NQT     6           // N-tiles per wave (6 x 16 = 96 columns)

// hex-kept taps of the 3x3 kernel: mask zeroes (0,2) and (2,0)
constexpr int TAP_R[NTAPS] = {0, 0, 1, 1, 1, 2, 2};
constexpr int TAP_C[NTAPS] = {0, 1, 0, 1, 2, 1, 2};

// float -> bf16 (round to nearest even)
__device__ __forceinline__ unsigned short f2bf(float f) {
    unsigned int u = __builtin_bit_cast(unsigned int, f);
    unsigned int r = u + 0x7FFFu + ((u >> 16) & 1u);
    return (unsigned short)(r >> 16);
}

// static 13x13 post-conv mask (true = keep)
__device__ __forceinline__ bool out_keep(int i, int j) {
    if (i <= 4 && j >= 8 + i) return false;
    if (i >= 2 && i <= 5 && j == 7 + i) return false;
    if (i >= 6 && i <= 8 && j == 12) return false;
    if (i >= 6 && j <= i - 6) return false;
    if (i >= 5 && i <= 7 && j == i - 5) return false;
    if ((i == 3 || i == 4) && j == 0) return false;
    return true;
}

// hexagonal wraparound copies on the padded 15x15 plane (dst <- src).
// e in [0,38): phase 1 (sources are untouched interior); [38,41): phase 2.
__device__ __forceinline__ void copy_pair(int e, int& dr, int& dc, int& sr, int& sc) {
    if (e < 7)       {               dr = 0;      dc = 1 + e;  sr = 12;     sc = 7 + e; }
    else if (e < 10) { int i = e-7;  dr = i;      dc = 8 + i;  sr = 6 + i;  sc = 2 + i; }
    else if (e < 14) { int i = e-10; dr = 3 + i;  dc = 10 + i; sr = 9 + i;  sc = 4 + i; }
    else if (e < 17) { int i = e-14; dr = 7 + i;  dc = 13;     sr = 1 + i;  sc = 1; }
    else if (e < 20) { int i = e-17; dr = 10 + i; dc = 14;     sr = 4 + i;  sc = 2; }
    else if (e < 28) { int i = e-20; dr = 13;     dc = 7 + i;  sr = 1;      sc = 1 + i; }
    else if (e < 32) { int i = e-28; dr = 9 + i;  dc = 3 + i;  sr = 3 + i;  sc = 9 + i; }
    else if (e < 34) { int i = e-32; dr = 4 + i;  dc = 1;      sr = 10 + i; sc = 13; }
    else if (e < 38) { int i = e-34; dr = i;      dc = 0;      sr = 6 + i;  sc = 12; }
    else             { int i = e-38; dr = 6 + i;  dc = 1 + i;  sr = i;      sc = 7 + i; }
}

// ---------------------------------------------------------------------------
// Prep: pack weights as wfrag[tap][mt][ks2][lane][j] in WMMA A-frag layout.
//   lane<16 : M = mt*16+lane,    kin=j      (0..15)
//   lane>=16: M = mt*16+lane-16, kin=16+j   (16..31)
//   ci = ks2*32 + kin
// ---------------------------------------------------------------------------
__global__ __launch_bounds__(256) void hex_prep_w(const float* __restrict__ w,
                                                  unsigned short* __restrict__ wfrag) {
    int f = blockIdx.x * 256 + threadIdx.x;          // 7*4*2*32*16 = 28672
    if (f >= NTAPS * 4 * 2 * 32 * 16) return;
    int j    = f & 15;
    int lane = (f >> 4) & 31;
    int ks2  = (f >> 9) & 1;
    int mt   = (f >> 10) & 3;
    int tap  = f >> 12;
    int m    = mt * 16 + (lane & 15);
    int kin  = (lane < 16) ? j : (16 + j);
    int ci   = ks2 * 32 + kin;
    int tt   = TAP_R[tap] * 3 + TAP_C[tap];
    wfrag[f] = f2bf(w[(m * CIN + ci) * 9 + tt]);
}

// ---------------------------------------------------------------------------
// Main: one workgroup per batch image, 8 waves (wave32).
// ---------------------------------------------------------------------------
__global__ __launch_bounds__(256, 1) void hexconv_wmma(
    const float* __restrict__ x, const float* __restrict__ bias,
    const unsigned short* __restrict__ wfrag, float* __restrict__ y) {

    __shared__ __align__(16) unsigned short xp[HP * HP * PSTRIDE]; // 32.4 KB

    const int tid = threadIdx.x;
    const int b   = blockIdx.x;

    // --- zero the padded image (16B stores) --------------------------------
    {
        U4 z; z.x = z.y = z.z = z.w = 0u;
        U4* xp4 = (U4*)xp;                      // 2025 x 16B
        for (int i = tid; i < HP * HP * PSTRIDE / 8; i += 256) xp4[i] = z;
    }
    __syncthreads();
    // --- interior fill: coalesced global read, bank-friendly LDS scatter ---
    const float* xb = x + (size_t)b * CIN * NPIX;
    for (int i = tid; i < CIN * NPIX; i += 256) {
        int ci = i / NPIX;
        int p  = i - ci * NPIX;
        int r  = p / HW, c = p - r * HW;
        xp[((r + 1) * HP + (c + 1)) * PSTRIDE + ci] = f2bf(xb[i]);
    }
    __syncthreads();
    // --- hex wraparound copies, phase 1 (independent) ----------------------
    for (int i = tid; i < 38 * CIN; i += 256) {
        int e = i >> 6, ci = i & 63;
        int dr, dc, sr, sc; copy_pair(e, dr, dc, sr, sc);
        xp[(dr * HP + dc) * PSTRIDE + ci] = xp[(sr * HP + sc) * PSTRIDE + ci];
    }
    __syncthreads();
    // --- phase 2 (reads (0,7) written in phase 1) --------------------------
    for (int i = tid; i < 3 * CIN; i += 256) {
        int e = 38 + (i >> 6), ci = i & 63;
        int dr, dc, sr, sc; copy_pair(e, dr, dc, sr, sc);
        xp[(dr * HP + dc) * PSTRIDE + ci] = xp[(sr * HP + sc) * PSTRIDE + ci];
    }
    __syncthreads();

    // --- implicit GEMM -----------------------------------------------------
    const int lane = tid & 31;
    const int wv   = tid >> 5;
    const int mt   = wv & 3;        // 4 M-tiles x 16 = 64 Cout
    const int ng   = wv >> 2;       // N wave-group
    const int l15  = lane & 15;
    const int hi   = lane >> 4;
    const int khE  = hi ? 16 : 0;   // element offset of this lane's K half

    const unsigned short* wl = wfrag + mt * 1024 + lane * 16;
    float* yb = y + (size_t)b * COUT * NPIX;

    float bv[8];
    #pragma unroll
    for (int v = 0; v < 8; ++v) bv[v] = bias[mt * 16 + v + hi * 8];

    // 6 N-tiles per wave: q -> columns (q>>1)*64 + ng*32 + (q&1)*16 + [0,16)
    int nQ[NQT], baseQ[NQT], rQ[NQT], cQ[NQT];
    #pragma unroll
    for (int q = 0; q < NQT; ++q) {
        int n = (q >> 1) * 64 + ng * 32 + (q & 1) * 16 + l15;
        int p = (n < NPIX) ? n : 0;
        int r = p / HW, c = p - r * HW;
        nQ[q] = n; rQ[q] = r; cQ[q] = c;
        baseQ[q] = (r * HP + c) * PSTRIDE + khE;   // window-origin element addr
    }

    v8f acc[NQT];
    #pragma unroll
    for (int q = 0; q < NQT; ++q) acc[q] = (v8f){};

    #pragma unroll
    for (int t = 0; t < NTAPS; ++t) {
        const int tofs = (TAP_R[t] * HP + TAP_C[t]) * PSTRIDE;     // imm
        #pragma unroll
        for (int ks2 = 0; ks2 < 2; ++ks2) {
            // A fragment: loaded exactly once per (tap, K-step), 32B/lane
            Frag a;
            const unsigned short* ap = wl + (t * 8 + ks2) * 512;
            a.q[0] = *(const U4*)(ap);
            a.q[1] = *(const U4*)(ap + 8);
            if (t < NTAPS - 1)
                __builtin_prefetch(ap + 8 * 512, 0, 3);

            // 6 B fragments: contiguous 32B LDS reads, compile-time offsets
            #pragma unroll
            for (int q = 0; q < NQT; ++q) {
                Frag bq;
                const unsigned short* bp = &xp[baseQ[q] + tofs + ks2 * 32];
                bq.q[0] = *(const U4*)(bp);
                bq.q[1] = *(const U4*)(bp + 8);
                acc[q] = __builtin_amdgcn_wmma_f32_16x16x32_bf16(
                             false, a.v, false, bq.v, (short)0, acc[q],
                             false, false);
            }
        }
    }

    // --- epilogue: bias + OUT_MASK, fp32 stores ----------------------------
    #pragma unroll
    for (int q = 0; q < NQT; ++q) {
        if (nQ[q] < NPIX) {
            float keep = out_keep(rQ[q], cQ[q]) ? 1.0f : 0.0f;
            #pragma unroll
            for (int v = 0; v < 8; ++v) {
                int co = mt * 16 + v + hi * 8;
                yb[co * NPIX + nQ[q]] = keep * (acc[q][v] + bv[v]);
            }
        }
    }
}

extern "C" void kernel_launch(void* const* d_in, const int* in_sizes, int n_in,
                              void* d_out, int out_size, void* d_ws, size_t ws_size,
                              hipStream_t stream) {
    const float* x    = (const float*)d_in[0];   // [4096,64,13,13]
    const float* w    = (const float*)d_in[1];   // [64,64,3,3]
    const float* bias = (const float*)d_in[2];   // [64]
    float* y          = (float*)d_out;           // [4096,64,13,13]
    unsigned short* wfrag = (unsigned short*)d_ws;   // 28672 bf16 = 56 KB

    hex_prep_w<<<112, 256, 0, stream>>>(w, wfrag);
    hexconv_wmma<<<B_TOTAL, 256, 0, stream>>>(x, bias, wfrag, y);
}
